// GATNet_46505905881183
// MI455X (gfx1250) — matrix operand order repaired and checked by reading, not compiled, not run on hardware
//
#include <hip/hip_runtime.h>
#include <hip/hip_bf16.h>
#include <stdint.h>

// ---------------------------------------------------------------------------
// CDNA5 / gfx1250 implementation of GATNet forward.
// Dense GEMMs: v_wmma_f32_16x16x32_f16, double-buffered LDS fed by
// global_load_async_to_lds_b128 (ASYNCcnt path). Requires ws >= ~300 MB.
// ---------------------------------------------------------------------------

typedef _Float16 half_t;
typedef __attribute__((ext_vector_type(16))) _Float16 v16h;
typedef __attribute__((ext_vector_type(8)))  float    v8f;
typedef __attribute__((ext_vector_type(4)))  uint32_t uint4v;

__device__ __forceinline__ unsigned enc_f(float f) {
  unsigned u = __float_as_uint(f);
  return (u >> 31) ? ~u : (u | 0x80000000u);   // monotonic float->uint map
}
__device__ __forceinline__ float dec_f(unsigned e) {
  return (e >> 31) ? __uint_as_float(e ^ 0x80000000u) : __uint_as_float(~e);
}

// LDS byte offset of a __shared__ object (flat LDS address truncates to
// addr[31:0] == LDS offset on gfx1250).
__device__ __forceinline__ uint32_t lds_off(const void* p) {
  return (uint32_t)(uintptr_t)p;
}
__device__ __forceinline__ void async_b128(uint32_t lds, const void* g) {
  asm volatile("global_load_async_to_lds_b128 %0, %1, off"
               :: "v"(lds), "v"((unsigned long long)(uintptr_t)g) : "memory");
}
__device__ __forceinline__ void async_b32(uint32_t lds, const void* g) {
  asm volatile("global_load_async_to_lds_b32 %0, %1, off"
               :: "v"(lds), "v"((unsigned long long)(uintptr_t)g) : "memory");
}
__device__ __forceinline__ void wait_async0() {
  asm volatile("s_wait_asynccnt 0x0" ::: "memory");
}

// ---------------------------------------------------------------------------
// Generic NT GEMM:  C[M,N] = act( A[M,K] * B[N,K]^T + bias[N] )
// A, B f16 row-major with leading dim == K (K multiple of 32).
// Block: 256 threads = 8 waves; tile 128(M) x 64(N); each wave: 32x32.
// Double-buffered LDS, async global->LDS staging. act: 0 = none, 1 = relu.
// ---------------------------------------------------------------------------
__global__ __launch_bounds__(256)
void gemm_nt_wmma(const half_t* __restrict__ A, const half_t* __restrict__ B,
                  const float* __restrict__ bias, float* __restrict__ C,
                  int M, int N, int K, int act)
{
  __shared__ half_t As[2][128][40];   // 32 halfs + pad, row = 80B (16B aligned)
  __shared__ half_t Bs[2][64][40];

  const int tid     = threadIdx.x;
  const int wave    = tid >> 5;
  const int lane    = tid & 31;
  const int half_id = lane >> 4;
  const int lr      = lane & 15;
  const int m0 = blockIdx.y * 128;
  const int n0 = blockIdx.x * 64;
  const int wm = (wave & 3) * 32;
  const int wn = (wave >> 2) * 32;

  v8f acc[2][2] = {};
  const uint4v zero4 = {0, 0, 0, 0};

  // stage one 32-wide K chunk into buffer `buf`:
  //   A: 128 rows x 32 halfs = 512 x 16B segments (2 per thread, async b128)
  //   B: 64 rows x 32 halfs  = 256 x 16B segments (1 per thread, async b128)
  auto stage = [&](int k0, int buf) {
#pragma unroll
    for (int s = 0; s < 2; ++s) {
      int seg  = tid + s * 256;
      int row  = seg >> 2;
      int part = (seg & 3) * 8;              // half offset within row
      int gr   = m0 + row;
      if (gr < M)
        async_b128(lds_off(&As[buf][row][part]),
                   A + (size_t)gr * K + k0 + part);
      else
        *(uint4v*)&As[buf][row][part] = zero4;   // zero-fill OOB rows
    }
    {
      int row  = tid >> 2;
      int part = (tid & 3) * 8;
      int gr   = n0 + row;
      if (gr < N)
        async_b128(lds_off(&Bs[buf][row][part]),
                   B + (size_t)gr * K + k0 + part);
      else
        *(uint4v*)&Bs[buf][row][part] = zero4;
    }
  };

  const int nchunks = K >> 5;
  stage(0, 0);

  for (int kc = 0; kc < nchunks; ++kc) {
    wait_async0();          // my async copies for chunk kc have landed
    __syncthreads();        // ... and everyone else's too
    if (kc + 1 < nchunks) stage((kc + 1) << 5, (kc + 1) & 1);
    const int buf = kc & 1;

    // A fragment: lane<16 -> K runs {0..7,16..23}; lane>=16 -> {8..15,24..31}
    const int ka0 = half_id ? 8 : 0;
    v16h a[2], b[2];
#pragma unroll
    for (int i = 0; i < 2; ++i) {
      const half_t* p = &As[buf][wm + i * 16 + lr][0];
      union { v16h h; uint4v u[2]; } t;
      t.u[0] = *(const uint4v*)(p + ka0);
      t.u[1] = *(const uint4v*)(p + ka0 + 16);
      a[i] = t.h;
    }
    // B fragment: col = lr, K runs = half_id ? {8..15,24..31} : {0..7,16..23}
#pragma unroll
    for (int j = 0; j < 2; ++j) {
      const half_t* p = &Bs[buf][wn + j * 16 + lr][half_id * 8];
      union { v16h h; uint4v u[2]; } t;
      t.u[0] = *(const uint4v*)p;
      t.u[1] = *(const uint4v*)(p + 16);
      b[j] = t.h;
    }
#pragma unroll
    for (int i = 0; i < 2; ++i)
#pragma unroll
      for (int j = 0; j < 2; ++j)
        acc[i][j] = __builtin_amdgcn_wmma_f32_16x16x32_f16(
            false, a[i], false, b[j], (short)0, acc[i][j], false, false);
    // no trailing barrier: next iteration's top barrier protects the buffers
  }

  // epilogue: C VGPR r -> rows r (lanes 0-15) / r+8 (lanes 16-31)
#pragma unroll
  for (int i = 0; i < 2; ++i) {
#pragma unroll
    for (int j = 0; j < 2; ++j) {
      int col = n0 + wn + j * 16 + lr;
#pragma unroll
      for (int r = 0; r < 8; ++r) {
        int row = m0 + wm + i * 16 + r + 8 * half_id;
        if (row < M && col < N) {
          float v = acc[i][j][r];
          if (bias) v += bias[col];
          if (act == 1) v = fmaxf(v, 0.0f);
          C[(size_t)row * N + col] = v;
        }
      }
    }
  }
}

// ---------------------------------------------------------------------------
// Conv over gathered embeddings, WMMA im2col-in-LDS (async staging).
// Eg: [B][1000][128] f16 (gathered embedding rows)
// Wf: [32][8000] f16  (Wc2 flattened, K index = c*8+k)
// Y : [B][32*121] f16, relu(conv + bc2), laid out as reshape(B, 32*121)
// One block per graph; M=128 padded positions, N=32, K=8000.
// ---------------------------------------------------------------------------
__global__ __launch_bounds__(256)
void conv2_wmma(const half_t* __restrict__ Eg, const half_t* __restrict__ Wf,
                const float* __restrict__ bc, half_t* __restrict__ Y)
{
  __shared__ half_t Es[4][144];    // 4 channels x 128 + zero pad tail
  __shared__ half_t Ws[32][40];

  const int b = blockIdx.x;
  const int tid = threadIdx.x, wave = tid >> 5, lane = tid & 31;
  const int half_id = lane >> 4, lr = lane & 15;
  const half_t* Eb = Eg + (size_t)b * 1000 * 128;
  v8f acc[2] = {};

  for (int cc = 0; cc < 1000; cc += 4) {
    {   // stage 4 embedding rows (4 x 128 halfs) via async b32
      int r   = tid >> 6;
      int idx = (tid & 63) * 2;
      async_b32(lds_off(&Es[r][idx]), Eb + (size_t)(cc + r) * 128 + idx);
      if ((tid & 63) < 16) Es[r][128 + (tid & 63)] = (half_t)0.0f;
    }
    if (tid < 64) {   // stage weight chunk (32 x 32 halfs) via async b128
      int o = tid >> 1, p = (tid & 1) * 16;
      const half_t* s = Wf + (size_t)o * 8000 + cc * 8 + p;
      async_b128(lds_off(&Ws[o][p]), s);
      async_b128(lds_off(&Ws[o][p + 8]), s + 8);
    }
    wait_async0();
    __syncthreads();

    // A fragment: A[pos, 8*c'+k] = Es[c'][pos + k]  (shifted-window im2col)
    int pp = wave * 16 + lr;
    int r0 = half_id ? 1 : 0;
    int r1 = r0 + 2;
    v16h a;
#pragma unroll
    for (int k = 0; k < 8; ++k) {
      a[k]     = Es[r0][pp + k];
      a[k + 8] = Es[r1][pp + k];
    }
#pragma unroll
    for (int j = 0; j < 2; ++j) {
      const half_t* p = &Ws[j * 16 + lr][half_id * 8];
      union { v16h h; uint4v u[2]; } t;
      t.u[0] = *(const uint4v*)p;
      t.u[1] = *(const uint4v*)(p + 16);
      acc[j] = __builtin_amdgcn_wmma_f32_16x16x32_f16(
          false, a, false, t.h, (short)0, acc[j], false, false);
    }
    __syncthreads();   // all reads done before next chunk's async writes
  }

#pragma unroll
  for (int j = 0; j < 2; ++j) {
    int o = j * 16 + lr;
    float bb = bc[o];
#pragma unroll
    for (int r = 0; r < 8; ++r) {
      int p = wave * 16 + r + 8 * half_id;
      if (p < 121) {
        float v = fmaxf(acc[j][r] + bb, 0.0f);
        Y[(size_t)b * 3872 + o * 121 + p] = (half_t)v;
      }
    }
  }
}

// ---------------------------------------------------------------------------
// Elementwise / graph kernels
// ---------------------------------------------------------------------------
__global__ void f32_to_f16_pad(const float* __restrict__ src,
                               half_t* __restrict__ dst, int R, int K, int Kp)
{
  int i = blockIdx.x * blockDim.x + threadIdx.x;
  if (i >= R * Kp) return;
  int k = i % Kp, r = i / Kp;
  dst[i] = (half_t)(k < K ? src[(size_t)r * K + k] : 0.0f);
}

__global__ void fill_u32(unsigned* dst, unsigned val, int n)
{
  int i = blockIdx.x * blockDim.x + threadIdx.x;
  if (i < n) dst[i] = val;
}

// per-node attention projections: s[n,h]=<h[n,h,:],a_src[h]>, d likewise
__global__ void gat_proj(const float* __restrict__ H,
                         const float* __restrict__ a_src,
                         const float* __restrict__ a_dst,
                         float* __restrict__ s, float* __restrict__ d,
                         int Nn, int Hh, int C)
{
  int i = blockIdx.x * blockDim.x + threadIdx.x;
  if (i >= Nn * Hh) return;
  int n = i / Hh, h = i % Hh;
  const float* row = H + (size_t)n * Hh * C + (size_t)h * C;
  float as = 0.0f, ad = 0.0f;
  for (int c = 0; c < C; ++c) {
    float v = row[c];
    as += v * a_src[h * C + c];
    ad += v * a_dst[h * C + c];
  }
  s[i] = as; d[i] = ad;
}

__global__ void gat_edge_max(const int* __restrict__ src,
                             const int* __restrict__ dst, int E, int Nn, int Hh,
                             const float* __restrict__ s,
                             const float* __restrict__ d,
                             unsigned* __restrict__ menc)
{
  int i = blockIdx.x * blockDim.x + threadIdx.x;
  int total = (E + Nn) * Hh;
  if (i >= total) return;
  int e = i / Hh, h = i % Hh;
  int sn = e < E ? src[e] : e - E;      // self loops appended
  int dn = e < E ? dst[e] : e - E;
  float v = s[sn * Hh + h] + d[dn * Hh + h];
  v = v > 0.0f ? v : 0.2f * v;          // leaky_relu(0.2)
  atomicMax(&menc[dn * Hh + h], enc_f(v));
}

__global__ void gat_edge_expsum(const int* __restrict__ src,
                                const int* __restrict__ dst, int E, int Nn,
                                int Hh, const float* __restrict__ s,
                                const float* __restrict__ d,
                                const unsigned* __restrict__ menc,
                                float* __restrict__ exbuf,
                                float* __restrict__ den)
{
  int i = blockIdx.x * blockDim.x + threadIdx.x;
  int total = (E + Nn) * Hh;
  if (i >= total) return;
  int e = i / Hh, h = i % Hh;
  int sn = e < E ? src[e] : e - E;
  int dn = e < E ? dst[e] : e - E;
  float v = s[sn * Hh + h] + d[dn * Hh + h];
  v = v > 0.0f ? v : 0.2f * v;
  float ex = __expf(v - dec_f(menc[dn * Hh + h]));
  exbuf[i] = ex;
  atomicAdd(&den[dn * Hh + h], ex);
}

__global__ void gat_edge_agg(const int* __restrict__ src,
                             const int* __restrict__ dst, int E, int Nn, int Hh,
                             int C, const float* __restrict__ H,
                             const float* __restrict__ exbuf,
                             const float* __restrict__ den,
                             float* __restrict__ out)
{
  int total = (E + Nn) * Hh * C;
  for (int i = blockIdx.x * blockDim.x + threadIdx.x; i < total;
       i += gridDim.x * blockDim.x) {
    int c = i % C;
    int t = i / C;
    int h = t % Hh;
    int e = t / Hh;
    int sn = e < E ? src[e] : e - E;
    int dn = e < E ? dst[e] : e - E;
    float alpha = exbuf[e * Hh + h] / den[dn * Hh + h];
    atomicAdd(&out[(size_t)dn * Hh * C + h * C + c],
              alpha * H[(size_t)sn * Hh * C + h * C + c]);
  }
}

// ELU(out + bias) -> f16, zero-padded to Kp columns (feeds next WMMA GEMM)
__global__ void elu_bias_f16_pad(const float* __restrict__ X,
                                 const float* __restrict__ bias,
                                 half_t* __restrict__ Y, int Nn, int K, int Kp)
{
  int i = blockIdx.x * blockDim.x + threadIdx.x;
  if (i >= Nn * Kp) return;
  int k = i % Kp, n = i / Kp;
  float v = 0.0f;
  if (k < K) {
    v = X[(size_t)n * K + k] + bias[k];
    v = v > 0.0f ? v : expm1f(v);
  }
  Y[i] = (half_t)v;
}

// global max pool of relu(out2 + b2) over sorted batch ids
__global__ void pool_max(const float* __restrict__ out2,
                         const float* __restrict__ b2,
                         const int* __restrict__ batch,
                         unsigned* __restrict__ xg_enc, int Nn, int C)
{
  int i = blockIdx.x * blockDim.x + threadIdx.x;
  if (i >= Nn * C) return;
  int n = i / C, c = i % C;
  float v = fmaxf(out2[i] + b2[c], 0.0f);
  atomicMax(&xg_enc[batch[n] * C + c], enc_f(v));
}

__global__ void xg_decode_f16(const unsigned* __restrict__ xg_enc,
                              half_t* __restrict__ xgA, int n)
{
  int i = blockIdx.x * blockDim.x + threadIdx.x;
  if (i < n) xgA[i] = (half_t)dec_f(xg_enc[i]);
}

// gather embedding rows to f16: E[b,c,h] = emb[t2[b,c], h]
__global__ void gather_emb(const int* __restrict__ t2,
                           const float* __restrict__ emb,
                           half_t* __restrict__ Eg, int total)
{
  int i = blockIdx.x * blockDim.x + threadIdx.x;
  if (i >= total) return;
  int h = i & 127;
  int r = i >> 7;
  Eg[i] = (half_t)emb[(size_t)t2[r] * 128 + h];
}

// small conv branch 1: relu(conv(target1, Wc1) + bc1) -> f16 [512][32*17]
__global__ void conv1_relu(const float* __restrict__ t1,
                           const float* __restrict__ W,
                           const float* __restrict__ bias,
                           half_t* __restrict__ Y)
{
  int i = blockIdx.x * blockDim.x + threadIdx.x;
  if (i >= 512 * 32 * 17) return;
  int p = i % 17;
  int t = i / 17;
  int o = t % 32;
  int b = t / 32;
  float acc = bias[o];
  for (int c = 0; c < 20; ++c)
    for (int k = 0; k < 8; ++k)
      acc += t1[(size_t)b * 480 + c * 24 + p + k] * W[o * 160 + c * 8 + k];
  Y[(size_t)b * 544 + o * 17 + p] = (half_t)fmaxf(acc, 0.0f);
}

__global__ void concat3_f16(const float* __restrict__ xg,
                            const float* __restrict__ xt1,
                            const float* __restrict__ xt2,
                            half_t* __restrict__ xc)
{
  int i = blockIdx.x * blockDim.x + threadIdx.x;
  if (i >= 512 * 384) return;
  int k = i % 384, b = i / 384;
  float v = (k < 128)   ? xg[b * 128 + k]
            : (k < 256) ? xt1[b * 128 + k - 128]
                        : xt2[b * 128 + k - 256];
  xc[i] = (half_t)v;
}

__global__ void final_dot(const float* __restrict__ fc2,
                          const float* __restrict__ Wo,
                          const float* __restrict__ bo,
                          float* __restrict__ out, int Bb, int K)
{
  int b = blockIdx.x * blockDim.x + threadIdx.x;
  if (b >= Bb) return;
  float acc = bo[0];
  for (int k = 0; k < K; ++k) acc += fc2[(size_t)b * K + k] * Wo[k];
  out[b] = acc;
}

// ---------------------------------------------------------------------------
extern "C" void kernel_launch(void* const* d_in, const int* in_sizes, int n_in,
                              void* d_out, int out_size, void* d_ws,
                              size_t ws_size, hipStream_t stream)
{
  const float* x    = (const float*)d_in[0];
  const int*   ei   = (const int*)d_in[1];
  const int*   batch= (const int*)d_in[2];
  const float* t1   = (const float*)d_in[3];
  const int*   t2   = (const int*)d_in[4];
  const float* W1   = (const float*)d_in[5];
  const float* a_s1 = (const float*)d_in[6];
  const float* a_d1 = (const float*)d_in[7];
  const float* b1   = (const float*)d_in[8];
  const float* W2   = (const float*)d_in[9];
  const float* a_s2 = (const float*)d_in[10];
  const float* a_d2 = (const float*)d_in[11];
  const float* b2   = (const float*)d_in[12];
  const float* Wg   = (const float*)d_in[13];
  const float* bg   = (const float*)d_in[14];
  const float* emb  = (const float*)d_in[15];
  const float* Wc2  = (const float*)d_in[16];
  const float* bc2  = (const float*)d_in[17];
  const float* Wc1  = (const float*)d_in[18];
  const float* bc1  = (const float*)d_in[19];
  const float* Wxt2 = (const float*)d_in[20];
  const float* bxt2 = (const float*)d_in[21];
  const float* Wxt1 = (const float*)d_in[22];
  const float* bxt1 = (const float*)d_in[23];
  const float* Wf1  = (const float*)d_in[24];
  const float* bf1  = (const float*)d_in[25];
  const float* Wf2  = (const float*)d_in[26];
  const float* bf2  = (const float*)d_in[27];
  const float* Wo   = (const float*)d_in[28];
  const float* bo   = (const float*)d_in[29];
  float* out = (float*)d_out;

  const int Nn = in_sizes[0] / 78;   // 30000
  const int E  = in_sizes[1] / 2;    // 240000
  const int Bb = 512;
  const int* esrc = ei;
  const int* edst = ei + E;

  char* ws = (char*)d_ws;

  // ---- big region: phase-overlapped buffers (stream order serializes) ----
  float*  h1    = (float*)(ws + 0);                 // [30000,780]  (GAT1 lin)
  float*  out1  = (float*)(ws + 100663296);         // [30000,780]  (GAT1 agg)
  half_t* h1act = (half_t*)(ws + 201326592);        // [30000,800]  f16 padded
  float*  h2    = (float*)(ws + 0);                 // [30000,128]  (reuses h1)
  float*  out2  = (float*)(ws + 16777216);          // [30000,128]
  half_t* E16   = (half_t*)(ws + 0);                // [512,1000,128] (after pool)
  half_t* xt2A  = (half_t*)(ws + 138412032);        // [512,3872]

  // ---- small persistent region (bump allocated) ----
  size_t off = 264241152ull;
  auto take = [&](size_t bytes) -> char* {
    char* p = ws + off;
    off += (bytes + 255) & ~255ull;
    return p;
  };
  half_t*  xA    = (half_t*)take((size_t)Nn * 96 * 2);
  half_t*  W1f   = (half_t*)take(780 * 96 * 2);
  half_t*  W2f   = (half_t*)take(128 * 800 * 2);
  half_t*  Wgf   = (half_t*)take(128 * 128 * 2);
  half_t*  Wc2f  = (half_t*)take(32 * 8000 * 2);
  half_t*  Wxt2f = (half_t*)take(128 * 3872 * 2);
  half_t*  Wxt1f = (half_t*)take(128 * 544 * 2);
  half_t*  Wf1f  = (half_t*)take(1024 * 384 * 2);
  half_t*  Wf2f  = (half_t*)take(512 * 1024 * 2);
  float*   s1    = (float*)take((size_t)Nn * 10 * 4);
  float*   d1    = (float*)take((size_t)Nn * 10 * 4);
  unsigned* menc1= (unsigned*)take((size_t)Nn * 10 * 4);
  float*   den1  = (float*)take((size_t)Nn * 10 * 4);
  float*   ex1   = (float*)take((size_t)(E + Nn) * 10 * 4);
  float*   s2    = (float*)take((size_t)Nn * 4);
  float*   d2    = (float*)take((size_t)Nn * 4);
  unsigned* menc2= (unsigned*)take((size_t)Nn * 4);
  float*   den2  = (float*)take((size_t)Nn * 4);
  float*   ex2   = (float*)take((size_t)(E + Nn) * 4);
  unsigned* xg_enc=(unsigned*)take(512 * 128 * 4);
  half_t*  xgA   = (half_t*)take(512 * 128 * 2);
  float*   xgw   = (float*)take(512 * 128 * 4);
  float*   xt2   = (float*)take(512 * 128 * 4);
  half_t*  xt1A  = (half_t*)take(512 * 544 * 2);
  float*   xt1   = (float*)take(512 * 128 * 4);
  half_t*  xcA   = (half_t*)take(512 * 384 * 2);
  float*   fc1   = (float*)take(512 * 1024 * 4);
  half_t*  fc1A  = (half_t*)take(512 * 1024 * 2);
  float*   fc2   = (float*)take(512 * 512 * 4);

  const int T = 256;
  auto nb = [&](long long t) { return (unsigned)((t + T - 1) / T); };
  auto gemm = [&](const half_t* A, const half_t* Bm, const float* bias,
                  float* C, int M, int N, int K, int act) {
    dim3 g((N + 63) / 64, (M + 127) / 128);
    gemm_nt_wmma<<<g, dim3(256), 0, stream>>>(A, Bm, bias, C, M, N, K, act);
  };

  // ---- weight / input conversions to f16 (K padded to mult of 32) ----
  f32_to_f16_pad<<<nb((long long)Nn * 96), T, 0, stream>>>(x, xA, Nn, 78, 96);
  f32_to_f16_pad<<<nb(780 * 96), T, 0, stream>>>(W1, W1f, 780, 78, 96);
  f32_to_f16_pad<<<nb(128 * 800), T, 0, stream>>>(W2, W2f, 128, 780, 800);
  f32_to_f16_pad<<<nb(128 * 128), T, 0, stream>>>(Wg, Wgf, 128, 128, 128);
  f32_to_f16_pad<<<nb(32 * 8000), T, 0, stream>>>(Wc2, Wc2f, 32, 8000, 8000);
  f32_to_f16_pad<<<nb(128 * 3872), T, 0, stream>>>(Wxt2, Wxt2f, 128, 3872, 3872);
  f32_to_f16_pad<<<nb(128 * 544), T, 0, stream>>>(Wxt1, Wxt1f, 128, 544, 544);
  f32_to_f16_pad<<<nb(1024 * 384), T, 0, stream>>>(Wf1, Wf1f, 1024, 384, 384);
  f32_to_f16_pad<<<nb(512 * 1024), T, 0, stream>>>(Wf2, Wf2f, 512, 1024, 1024);

  // ---- GAT layer 1 ----
  gemm(xA, W1f, nullptr, h1, Nn, 780, 96, 0);                     // h1 = x@W1^T
  gat_proj<<<nb((long long)Nn * 10), T, 0, stream>>>(h1, a_s1, a_d1, s1, d1,
                                                     Nn, 10, 78);
  hipMemsetAsync(menc1, 0, (size_t)Nn * 10 * 4, stream);
  hipMemsetAsync(den1, 0, (size_t)Nn * 10 * 4, stream);
  gat_edge_max<<<nb((long long)(E + Nn) * 10), T, 0, stream>>>(
      esrc, edst, E, Nn, 10, s1, d1, menc1);
  gat_edge_expsum<<<nb((long long)(E + Nn) * 10), T, 0, stream>>>(
      esrc, edst, E, Nn, 10, s1, d1, menc1, ex1, den1);
  hipMemsetAsync(out1, 0, (size_t)Nn * 780 * 4, stream);
  gat_edge_agg<<<16384, T, 0, stream>>>(esrc, edst, E, Nn, 10, 78, h1, ex1,
                                        den1, out1);
  elu_bias_f16_pad<<<nb((long long)Nn * 800), T, 0, stream>>>(out1, b1, h1act,
                                                              Nn, 780, 800);

  // ---- GAT layer 2 ----
  gemm(h1act, W2f, nullptr, h2, Nn, 128, 800, 0);
  gat_proj<<<nb((long long)Nn), T, 0, stream>>>(h2, a_s2, a_d2, s2, d2, Nn, 1,
                                                128);
  hipMemsetAsync(menc2, 0, (size_t)Nn * 4, stream);
  hipMemsetAsync(den2, 0, (size_t)Nn * 4, stream);
  gat_edge_max<<<nb((long long)(E + Nn)), T, 0, stream>>>(esrc, edst, E, Nn, 1,
                                                          s2, d2, menc2);
  gat_edge_expsum<<<nb((long long)(E + Nn)), T, 0, stream>>>(
      esrc, edst, E, Nn, 1, s2, d2, menc2, ex2, den2);
  hipMemsetAsync(out2, 0, (size_t)Nn * 128 * 4, stream);
  gat_edge_agg<<<16384, T, 0, stream>>>(esrc, edst, E, Nn, 1, 128, h2, ex2,
                                        den2, out2);

  // ---- global max pool + Wg ----
  fill_u32<<<nb(512 * 128), T, 0, stream>>>(xg_enc, 0x007FFFFFu, 512 * 128);
  pool_max<<<nb((long long)Nn * 128), T, 0, stream>>>(out2, b2, batch, xg_enc,
                                                      Nn, 128);
  xg_decode_f16<<<nb(512 * 128), T, 0, stream>>>(xg_enc, xgA, 512 * 128);
  gemm(xgA, Wgf, bg, xgw, 512, 128, 128, 1);                      // relu

  // ---- protein branch 2 (big conv, reuses graph-phase scratch) ----
  gather_emb<<<nb((long long)512 * 1000 * 128), T, 0, stream>>>(
      t2, emb, E16, 512 * 1000 * 128);
  conv2_wmma<<<512, 256, 0, stream>>>(E16, Wc2f, bc2, xt2A);
  gemm(xt2A, Wxt2f, bxt2, xt2, 512, 128, 3872, 0);

  // ---- protein branch 1 (tiny conv) ----
  conv1_relu<<<nb(512 * 32 * 17), T, 0, stream>>>(t1, Wc1, bc1, xt1A);
  gemm(xt1A, Wxt1f, bxt1, xt1, 512, 128, 544, 0);

  // ---- head ----
  concat3_f16<<<nb(512 * 384), T, 0, stream>>>(xgw, xt1, xt2, xcA);
  gemm(xcA, Wf1f, bf1, fc1, 512, 1024, 384, 1);                   // relu
  f32_to_f16_pad<<<nb(512 * 1024), T, 0, stream>>>(fc1, fc1A, 512, 1024, 1024);
  gemm(fc1A, Wf2f, bf2, fc2, 512, 512, 1024, 1);                  // relu
  final_dot<<<nb(512), T, 0, stream>>>(fc2, Wo, bo, out, Bb, 512);

  (void)n_in; (void)out_size; (void)ws_size;
}
